// RNN_90323162235592
// MI455X (gfx1250) — matrix operand order repaired and checked by reading, not compiled, not run on hardware
//
#include <hip/hip_runtime.h>
#include <hip/hip_bf16.h>
#include <stdint.h>

#define SEQ   512
#define BATCH 256
#define DIM   512
#define HID   512

typedef __attribute__((ext_vector_type(16))) __bf16 v16bf;
typedef __attribute__((ext_vector_type(8)))  float  v8f;
typedef __attribute__((ext_vector_type(4)))  float  v4f;

union Frag16 { v16bf v; uint32_t d[8]; uint4 q[2]; };

__device__ __forceinline__ uint32_t pack2bf(float lo, float hi) {
    union { __bf16 h[2]; uint32_t u; } p;
    p.h[0] = (__bf16)lo;
    p.h[1] = (__bf16)hi;
    return p.u;
}

// ---------------------------------------------------------------------------
// CDNA5 async global->LDS copy (ASYNCcnt-tracked), per ISA 08_async_tensor §4.
// ---------------------------------------------------------------------------
__device__ __forceinline__ void async_b128(uint32_t lds_addr, const void* gsrc) {
    asm volatile("global_load_async_to_lds_b128 %0, %1, off"
                 :: "v"(lds_addr), "v"(gsrc)
                 : "memory");
}
#define WAIT_ASYNC(N) asm volatile("s_wait_asynccnt %0" :: "i"(N) : "memory")

// ---------------------------------------------------------------------------
// Fragment helpers
// ---------------------------------------------------------------------------

// Synchronous 4-fragment load from the tiled weight buffer (2x b128 each).
__device__ __forceinline__ void load_b4(Frag16 b[4], const uint32_t* __restrict__ wb,
                                        int jt0, int kc, int lane) {
#pragma unroll
    for (int i = 0; i < 4; ++i) {
        const uint32_t* bp = wb + (size_t)((jt0 + i) * 16 + kc) * 256u + lane * 8;
        b[i].q[0] = *(const uint4*)(bp);
        b[i].q[1] = *(const uint4*)(bp + 4);
    }
}

// Stage the 4 B-fragments of K-chunk kc (4KB, identity layout) into LDS with
// 8 async b128 ops (each: 32 lanes x 16B).
__device__ __forceinline__ void stage_b(uint32_t ldsBase, const uint32_t* __restrict__ wb,
                                        int jt0, int kc, int lane) {
#pragma unroll
    for (int j = 0; j < 8; ++j) {
        int i  = j >> 1;     // fragment 0..3
        int hh = j & 1;      // 512B half of the 1KB fragment
        const uint32_t* g = wb + (size_t)((jt0 + i) * 16 + kc) * 256u + hh * 128 + lane * 4;
        async_b128(ldsBase + (uint32_t)(j * 512 + lane * 16), g);
    }
}

// A-fragment from a bf16 row-major LDS tile (row stride 260 dwords):
// two ds_load_b128 (16B-aligned).
__device__ __forceinline__ void load_a_h(Frag16& a, const uint32_t* hrow,
                                         int kc, int half) {
    a.q[0] = *(const uint4*)(hrow + kc * 16 + half * 8);
    a.q[1] = *(const uint4*)(hrow + kc * 16 + half * 8 + 4);
}

#define WMMA_BF16(A, B, C) \
    __builtin_amdgcn_wmma_f32_16x16x32_bf16(false, (A), false, (B), (short)0, (C), false, false)

// ---------------------------------------------------------------------------
// Phase 0: repack W_hx / W_hh (fp32 row-major) into bf16 B-fragment tiles.
// ---------------------------------------------------------------------------
__global__ void convert_w_kernel(const float* __restrict__ Whx,
                                 const float* __restrict__ Whh,
                                 uint32_t* __restrict__ wb) {
    int f   = blockIdx.x;
    int mat = f >> 9;
    int fi  = f & 511;
    int jt  = fi >> 4;
    int kc  = fi & 15;
    const float* W = mat ? Whh : Whx;

    int lane = threadIdx.x;
    int n    = lane & 15;
    int half = lane >> 4;
    int j    = jt * 16 + n;

    uint32_t o[8];
#pragma unroll
    for (int v = 0; v < 8; ++v) {
        int k = kc * 32 + half * 16 + 2 * v;
        o[v] = pack2bf(W[(size_t)j * DIM + k], W[(size_t)j * DIM + k + 1]);
    }
    uint32_t* dst = wb + (size_t)mat * (512u * 256u) + (size_t)fi * 256u + lane * 8;
    *(uint4*)(dst)     = make_uint4(o[0], o[1], o[2], o[3]);
    *(uint4*)(dst + 4) = make_uint4(o[4], o[5], o[6], o[7]);
}

// ---------------------------------------------------------------------------
// Phase 1: xp = x @ W_hx^T + b_hx.  Each WG owns 128 rows and ALL 512 output
// columns: x is read from HBM exactly once (NT loads), converted to bf16 into
// LDS once, then 8 column blocks are computed reading A from LDS and B
// (double-buffered) from the L2-resident tiled W_hx. xp stores are NT
// (streamed, no reuse before the scan).
// ---------------------------------------------------------------------------
__global__ __launch_bounds__(256)
void xp_gemm_kernel(const float* __restrict__ x,
                    const uint32_t* __restrict__ wbx,
                    const float* __restrict__ bhx,
                    float* __restrict__ xp) {
    __shared__ __align__(16) uint32_t xS[8][16 * 260];   // per-wave 16x520 bf16

    const int lane = threadIdx.x & 31;
    const int wave = threadIdx.x >> 5;
    const int n    = lane & 15;
    const int half = lane >> 4;
    const int m0   = (blockIdx.x * 8 + wave) * 16;

    // Stage this wave's 16x512 fp32 tile as bf16 row-major (wave-private:
    // no barrier needed; dscnt ordering handles the staging->use dependency).
    uint32_t* xw = &xS[wave][0];
    for (int r = 0; r < 16; ++r) {
        const v4f* src = (const v4f*)(x + (size_t)(m0 + r) * DIM) + lane * 4;
        v4f f0 = __builtin_nontemporal_load(src + 0);
        v4f f1 = __builtin_nontemporal_load(src + 1);
        v4f f2 = __builtin_nontemporal_load(src + 2);
        v4f f3 = __builtin_nontemporal_load(src + 3);
        uint32_t* dst = xw + r * 260 + lane * 8;
        uint4 q0 = make_uint4(pack2bf(f0[0], f0[1]), pack2bf(f0[2], f0[3]),
                              pack2bf(f1[0], f1[1]), pack2bf(f1[2], f1[3]));
        uint4 q1 = make_uint4(pack2bf(f2[0], f2[1]), pack2bf(f2[2], f2[3]),
                              pack2bf(f3[0], f3[1]), pack2bf(f3[2], f3[3]));
        *(uint4*)(dst)     = q0;
        *(uint4*)(dst + 4) = q1;
    }

    const uint32_t* arow = xw + n * 260;

    for (int jb = 0; jb < 8; ++jb) {
        const int jt0 = jb * 4;
        v8f acc[4] = {};
        Frag16 bA[4], bB[4], a0, a1;

        load_b4(bA, wbx, jt0, 0, lane);
        for (int kc = 0; kc < 16; kc += 2) {
            load_b4(bB, wbx, jt0, kc + 1, lane);
            load_a_h(a0, arow, kc, half);
#pragma unroll
            for (int i = 0; i < 4; ++i) acc[i] = WMMA_BF16(a0.v, bA[i].v, acc[i]);

            if (kc + 2 < 16) load_b4(bA, wbx, jt0, kc + 2, lane);
            load_a_h(a1, arow, kc + 1, half);
#pragma unroll
            for (int i = 0; i < 4; ++i) acc[i] = WMMA_BF16(a1.v, bB[i].v, acc[i]);
        }

#pragma unroll
        for (int i = 0; i < 4; ++i) {
            int col   = jb * 64 + i * 16 + n;
            float bia = bhx[col];
#pragma unroll
            for (int r = 0; r < 8; ++r) {
                int row = m0 + r + half * 8;
                __builtin_nontemporal_store(acc[i][r] + bia, &xp[(size_t)row * HID + col]);
            }
        }
    }
}

// ---------------------------------------------------------------------------
// Phase 2: persistent scan. 16 WGs x 8 waves. h in LDS (bf16, padded rows).
// W_hh fragments stream L2 -> LDS via async b128 copies, double-buffered per
// K-chunk; WMMAs consume via ds_load_b128. 64 WMMAs/wave/step.
// ---------------------------------------------------------------------------
__global__ __launch_bounds__(256)
void rnn_scan_kernel(const uint32_t* __restrict__ wbh,
                     const float* __restrict__ bhh,
                     const float* __restrict__ xp,
                     float* __restrict__ out) {
    __shared__ __align__(16) uint32_t hLds[16 * 260];     // 16 x 520 bf16
    __shared__ __align__(16) uint32_t bS[8][2][1024];     // per-wave 2 x 4KB staging

    const int lane = threadIdx.x & 31;
    const int wave = threadIdx.x >> 5;
    const int n    = lane & 15;
    const int half = lane >> 4;
    const int b0   = blockIdx.x * 16;
    const int n0   = wave * 64;
    const int jt0  = n0 >> 4;

    for (int i = threadIdx.x; i < 16 * 260; i += 256) hLds[i] = 0u;

    float bias[4];
#pragma unroll
    for (int i = 0; i < 4; ++i) bias[i] = bhh[n0 + i * 16 + n];

    const uint32_t* hrow = hLds + n * 260;
    // LDS byte address of this wave's staging buffers (ISA: low 32 bits of a
    // generic LDS pointer are the LDS byte offset).
    const uint32_t bBase = (uint32_t)(uintptr_t)(&bS[wave][0][0]);
    float hout[4][8];

    for (int t = 0; t < SEQ; ++t) {
        __syncthreads();               // h writes from step t-1 visible

        // Pull next step's xp slice toward caches (global_prefetch_b8).
        if (t + 1 < SEQ) {
            const char* pf = (const char*)(xp + ((size_t)(t + 1) * BATCH + b0) * HID);
            __builtin_prefetch(pf + (size_t)threadIdx.x * 128, 0, 0);
        }

        v8f acc[4] = {};
        stage_b(bBase, wbh, jt0, 0, lane);            // chunk 0 -> buf 0

#pragma unroll
        for (int kc = 0; kc < 16; ++kc) {
            const int cur = kc & 1;
            if (kc + 1 < 16) {
                stage_b(bBase + (uint32_t)((cur ^ 1) * 4096), wbh, jt0, kc + 1, lane);
                WAIT_ASYNC(8);         // in-order: current chunk has landed
            } else {
                WAIT_ASYNC(0);
            }

            Frag16 a;
            load_a_h(a, hrow, kc, half);
            const uint32_t* bbuf = &bS[wave][cur][0];
#pragma unroll
            for (int i = 0; i < 4; ++i) {
                Frag16 b;
                b.q[0] = *(const uint4*)(bbuf + i * 256 + lane * 8);
                b.q[1] = *(const uint4*)(bbuf + i * 256 + lane * 8 + 4);
                acc[i] = WMMA_BF16(a.v, b.v, acc[i]);
            }
        }
        __syncthreads();               // all reads of h done before overwrite

        const float* xpt = xp + ((size_t)t * BATCH + b0) * HID;
        __bf16* hp = (__bf16*)hLds;
#pragma unroll
        for (int i = 0; i < 4; ++i) {
            int col = n0 + i * 16 + n;
#pragma unroll
            for (int r = 0; r < 8; ++r) {
                int row = r + half * 8;
                float v = acc[i][r] + xpt[(size_t)row * HID + col] + bias[i];
                float h = tanhf(v);
                hout[i][r] = h;
                hp[row * 520 + col] = (__bf16)h;
            }
        }
    }

#pragma unroll
    for (int i = 0; i < 4; ++i) {
        int col = n0 + i * 16 + n;
#pragma unroll
        for (int r = 0; r < 8; ++r) {
            int row = b0 + r + half * 8;
            out[(size_t)row * HID + col] = hout[i][r];
        }
    }
}

// ---------------------------------------------------------------------------
extern "C" void kernel_launch(void* const* d_in, const int* in_sizes, int n_in,
                              void* d_out, int out_size, void* d_ws, size_t ws_size,
                              hipStream_t stream) {
    const float* x   = (const float*)d_in[0];
    const float* Whx = (const float*)d_in[1];
    const float* bhx = (const float*)d_in[2];
    const float* Whh = (const float*)d_in[3];
    const float* bhh = (const float*)d_in[4];
    float* out = (float*)d_out;

    // ws layout: [0, 512KB) tiled bf16 W_hx | [512KB, 1MB) tiled bf16 W_hh
    //            [1MB, 1MB+256MB) xp fp32 [SEQ][BATCH][HID]
    uint32_t* wb  = (uint32_t*)d_ws;
    uint32_t* wbh = wb + 512u * 256u;
    float*    xp  = (float*)((char*)d_ws + 2u * 512u * 1024u);

    convert_w_kernel<<<dim3(1024), dim3(32), 0, stream>>>(Whx, Whh, wb);
    xp_gemm_kernel<<<dim3((SEQ * BATCH) / 128), dim3(256), 0, stream>>>(
        x, wb, bhx, xp);
    rnn_scan_kernel<<<dim3(BATCH / 16), dim3(256), 0, stream>>>(wbh, bhh, xp, out);
}
